// LocalFeatureAggregation_59201829208478
// MI455X (gfx1250) — compile-verified, matrix operand order checked by reading, not run on hardware
//
#include <hip/hip_runtime.h>
#include <hip/hip_bf16.h>
#include <cstddef>

// ---------------------------------------------------------------------------
// LocalFeatureAggregation (RandLA-Net style) for MI455X / gfx1250 (wave32).
// B=2, N=16384, K=16, D_IN=128, D_OUT=256, GROUPS=16, EPS=1e-6.
// Dense stages: v_wmma_f32_16x16x32_bf16, pre-converted bf16 weights,
// bf16-transposed LDS tiles, A-fragment reused across 4 point tiles.
// ---------------------------------------------------------------------------

#define NPTS 16384
#define NB   2
#define KNN  16

typedef __attribute__((ext_vector_type(16))) __bf16 v16bf;
typedef __attribute__((ext_vector_type(8)))  __bf16 v8bf;
typedef __attribute__((ext_vector_type(8)))  float  v8f;

__device__ __forceinline__ unsigned packbf2(float a, float b) {
    union { __bf16 h[2]; unsigned u; } c;
    c.h[0] = (__bf16)a; c.h[1] = (__bf16)b;
    return c.u;
}

// Row of PT 16x16 output tiles: acc[p] = W(16 x CIN) * X_p(CIN x 16).
// A-fragment loaded once per K-step and reused for all PT B-tiles.
//   Wbf: bf16 weights, row-major, row stride CIN (offset to tile row o0).
//   sXT: LDS bf16 tile, transposed [PT*16 cols][CIN+8] (pad avoids conflicts).
template<int CIN, int PT>
__device__ __forceinline__ void wmma_row(const __bf16* __restrict__ Wbf,
                                         const __bf16* sXT, int lane, v8f* acc) {
    const int mrow = lane & 15;          // A row
    const int kb   = (lane >> 4) << 4;   // K half: 0 or 16
    const int j    = lane & 15;          // B column within 16-pt tile
    const __bf16* wrow = Wbf + mrow * CIN + kb;
    v8f zero = {};
#pragma unroll
    for (int p = 0; p < PT; ++p) acc[p] = zero;
    for (int kk = 0; kk < CIN; kk += 32) {
        const v8bf* ap = (const v8bf*)(wrow + kk);
        v8bf a0 = ap[0], a1 = ap[1];
        v16bf a = __builtin_shufflevector(a0, a1, 0,1,2,3,4,5,6,7,8,9,10,11,12,13,14,15);
#pragma unroll
        for (int p = 0; p < PT; ++p) {
            const v8bf* bp = (const v8bf*)(sXT + (p * 16 + j) * (CIN + 8) + kb + kk);
            v8bf b0 = bp[0], b1 = bp[1];
            v16bf bb = __builtin_shufflevector(b0, b1, 0,1,2,3,4,5,6,7,8,9,10,11,12,13,14,15);
            acc[p] = __builtin_amdgcn_wmma_f32_16x16x32_bf16(
                false, a, false, bb, (short)0, acc[p], false, false);
        }
    }
}

// Stage a CIN x TW fp32 channel-major global tile into transposed bf16 LDS.
template<int CIN, int TW>
__device__ __forceinline__ void stage_tile(const float* __restrict__ X, int n0,
                                           __bf16* sXT, int tid) {
    for (int i = tid; i < (CIN / 2) * TW; i += 256) {
        int c2 = (i / TW) * 2, j = i % TW;
        float f0 = X[(size_t)c2 * NPTS + n0 + j];
        float f1 = X[(size_t)(c2 + 1) * NPTS + n0 + j];
        *(unsigned*)(sXT + j * (CIN + 8) + c2) = packbf2(f0, f1);
    }
}

// --------------------------- small helpers ---------------------------------

__global__ void k_zero(float* ws) {
    for (int t = threadIdx.x; t < 1024; t += 256) ws[t] = 0.f;
}

__global__ void k_cvt_bf16(const float* __restrict__ src, __bf16* __restrict__ dst, int count) {
    int i2 = (blockIdx.x * 256 + threadIdx.x) * 2;
    if (i2 < count) *(unsigned*)(dst + i2) = packbf2(src[i2], src[i2 + 1]);
}

// sum/ssq (B*16 cells) -> mean, inv_std
__global__ void k_stat_fin(const float* sum, const float* ssq, float* mu, float* inv, float cnt) {
    int i = threadIdx.x;
    if (i < NB * 16) {
        float m = sum[i] / cnt;
        float v = ssq[i] / cnt - m * m;
        mu[i]  = m;
        inv[i] = rsqrtf(v + 1e-6f);
    }
}

// ------------- stage 1: conv1 (128x128) + leaky 0.2, 64-pt tiles -----------

__global__ void k_conv1(const float* __restrict__ F, const __bf16* __restrict__ Wbf,
                        const float* __restrict__ Bb, float* __restrict__ X1) {
    const int b = blockIdx.y, n0 = blockIdx.x * 64, tid = threadIdx.x;
    __shared__ alignas(16) __bf16 sXT[64 * 136];
    stage_tile<128, 64>(F + (size_t)b * 128 * NPTS, n0, sXT, tid);
    __syncthreads();
    const int wv = tid >> 5, lane = tid & 31;
    const int mb = (lane >> 4) << 3, j = lane & 15;
    v8f acc[4];
    wmma_row<128, 4>(Wbf + (size_t)wv * 16 * 128, sXT, lane, acc);
#pragma unroll
    for (int p = 0; p < 4; ++p)
#pragma unroll
        for (int r = 0; r < 8; ++r) {
            int o = wv * 16 + mb + r;
            float v = acc[p][r] + Bb[o];
            X1[((size_t)b * 128 + o) * NPTS + n0 + p * 16 + j] = v > 0.f ? v : 0.2f * v;
        }
}

// -------- stage 2: per-batch 10x10 geo second moments + first moments -------
// geo = [ctr(3), nbr(3), ctr-nbr(3), dist]; exact GN stats of W*geo+b follow.

__global__ void k_geomom(const float* __restrict__ coords, const int* __restrict__ kidx,
                         const float* __restrict__ kdist, float* __restrict__ mom) {
    const int b = blockIdx.y, tid = threadIdx.x;
    const float* cb = coords + (size_t)b * NPTS * 3;
    float accM[55], accS[10];
#pragma unroll
    for (int i = 0; i < 55; ++i) accM[i] = 0.f;
#pragma unroll
    for (int i = 0; i < 10; ++i) accS[i] = 0.f;
    const int base = blockIdx.x * 4096;
    for (int it = 0; it < 16; ++it) {
        int e = base + it * 256 + tid;   // e in [0, N*K)
        int n = e >> 4, k = e & 15;
        float cx = cb[n * 3], cy = cb[n * 3 + 1], cz = cb[n * 3 + 2];
        int id = kidx[((size_t)b * NPTS + n) * KNN + k];
        float nx = cb[id * 3], ny = cb[id * 3 + 1], nz = cb[id * 3 + 2];
        float dd = kdist[((size_t)b * NPTS + n) * KNN + k];
        float geo[10] = {cx, cy, cz, nx, ny, nz, cx - nx, cy - ny, cz - nz, dd};
        int idx = 0;
#pragma unroll
        for (int jj = 0; jj < 10; ++jj) {
            accS[jj] += geo[jj];
#pragma unroll
            for (int j2 = jj; j2 < 10; ++j2) { accM[idx] += geo[jj] * geo[j2]; ++idx; }
        }
    }
    __shared__ float red[65];
    if (tid < 65) red[tid] = 0.f;
    __syncthreads();
#pragma unroll
    for (int i = 0; i < 55; ++i) atomicAdd(&red[i], accM[i]);
#pragma unroll
    for (int i = 0; i < 10; ++i) atomicAdd(&red[55 + i], accS[i]);
    __syncthreads();
    if (tid < 65) atomicAdd(&mom[b * 65 + tid], red[tid]);
}

// Analytic GN stats for both LSE convs (g = W*geo + b, linear in geo).
__global__ void k_lse_stats(const float* __restrict__ mom,
                            const float* __restrict__ w1_, const float* __restrict__ b1_,
                            const float* __restrict__ w2_, const float* __restrict__ b2_,
                            float* __restrict__ lmu, float* __restrict__ linv) {
    const int tid = threadIdx.x;
    if (tid >= 64) return;
    const int b = tid >> 5, conv = (tid >> 4) & 1, g = tid & 15;
    const float* W  = conv ? w2_ : w1_;
    const float* Bb = conv ? b2_ : b1_;
    const float* M = mom + b * 65;
    const float* S = M + 55;
    const float NK = (float)NPTS * (float)KNN;
    float sum = 0.f, ss = 0.f;
    for (int oo = 0; oo < 8; ++oo) {
        int o = g * 8 + oo;
        float wv[10];
#pragma unroll
        for (int j = 0; j < 10; ++j) wv[j] = W[o * 10 + j];
        float q = 0.f, sd = 0.f;
        int idx = 0;
#pragma unroll
        for (int jj = 0; jj < 10; ++jj) {
            sd += wv[jj] * S[jj];
#pragma unroll
            for (int j2 = jj; j2 < 10; ++j2) {
                float c = wv[jj] * wv[j2] * M[idx];
                q += (j2 == jj) ? c : 2.f * c;
                ++idx;
            }
        }
        float bo = Bb[o];
        sum += sd + NK * bo;
        ss  += q + 2.f * bo * sd + NK * bo * bo;
    }
    float cnt = 8.f * NK;
    float mean = sum / cnt;
    float var  = ss / cnt - mean * mean;
    int oi = (conv * NB + b) * 16 + g;
    lmu[oi]  = mean;
    linv[oi] = rsqrtf(var + 1e-6f);
}

// ----- fused LSE + attentive-pool conv: build m(256 x 16pts) in LDS, WMMA ---
// m[0:128]   = mean_k relu(GN(lse_w * geo))   (stats precomputed analytically)
// m[128:256] = mean_k NF[:, idx[n,k]]         (gather from L2-resident map)
// Phase-2 threads own 8 contiguous channels of one column -> pack K-means
// straight into the transposed bf16 WMMA tile (no fp32 staging, no atomics).

template<int COUT, int GSHIFT>
__global__ void k_pool_a(const float* __restrict__ coords, const int* __restrict__ kidx,
                         const float* __restrict__ kdist,
                         const float* __restrict__ lw, const float* __restrict__ lb,
                         const float* __restrict__ lgw, const float* __restrict__ lgb,
                         const float* __restrict__ lmu, const float* __restrict__ linv,
                         const float* __restrict__ NF,
                         const __bf16* __restrict__ PWbf, const float* __restrict__ PB,
                         float* __restrict__ T, float* __restrict__ gsum, float* __restrict__ gssq) {
    const int b = blockIdx.y, n0 = blockIdx.x * 16, tid = threadIdx.x;
    __shared__ float sGeo[16][16][10];
    __shared__ int   sId[16][16];
    __shared__ alignas(16) __bf16 sMT[16 * 264];
    __shared__ float sLW[1280], sLB[128], sGW[128], sGB[128];
    __shared__ float sMU[16], sINV[16], sGS[16], sGSS[16];
    for (int i = tid; i < 1280; i += 256) sLW[i] = lw[i];
    if (tid < 128) { sLB[tid] = lb[tid]; sGW[tid] = lgw[tid]; sGB[tid] = lgb[tid]; }
    if (tid < 16)  { sMU[tid] = lmu[b * 16 + tid]; sINV[tid] = linv[b * 16 + tid];
                     sGS[tid] = 0.f; sGSS[tid] = 0.f; }
    // phase 1: thread=(point p, neighbor k) -> geo vector + neighbor id to LDS
    {
        const int p = tid >> 4, k = tid & 15, n = n0 + p;
        const float* cb = coords + (size_t)b * NPTS * 3;
        float cx = cb[n * 3], cy = cb[n * 3 + 1], cz = cb[n * 3 + 2];
        int id = kidx[((size_t)b * NPTS + n) * KNN + k];
        float nx = cb[id * 3], ny = cb[id * 3 + 1], nz = cb[id * 3 + 2];
        float dd = kdist[((size_t)b * NPTS + n) * KNN + k];
        sId[p][k] = id;
        float geo[10] = {cx, cy, cz, nx, ny, nz, cx - nx, cy - ny, cz - nz, dd};
#pragma unroll
        for (int j = 0; j < 10; ++j) sGeo[p][k][j] = geo[j];
    }
    __syncthreads();
    // phase 2: thread=(point p, channel-chunk oc): private K-mean, direct
    // packed bf16 stores into sMT column p. Group of channel o = o>>3 == oc.
    {
        const int p = tid >> 4, oc = tid & 15;
        float macc[8];
#pragma unroll
        for (int r = 0; r < 8; ++r) macc[r] = 0.f;
        for (int k = 0; k < KNN; ++k) {
            float g10[10];
#pragma unroll
            for (int j = 0; j < 10; ++j) g10[j] = sGeo[p][k][j];
#pragma unroll
            for (int r = 0; r < 8; ++r) {
                int o = oc * 8 + r;
                float g = sLB[o];
#pragma unroll
                for (int j = 0; j < 10; ++j) g += sLW[o * 10 + j] * g10[j];
                float v = (g - sMU[oc]) * sINV[oc] * sGW[o] + sGB[o];
                macc[r] += v > 0.f ? v : 0.f;
            }
        }
        v8bf pk;
#pragma unroll
        for (int r = 0; r < 8; ++r) pk[r] = (__bf16)(macc[r] * 0.0625f);
        *(v8bf*)(sMT + p * 264 + oc * 8) = pk;

        const float* nfb = NF + (size_t)b * 128 * NPTS;
        float nacc[8];
#pragma unroll
        for (int r = 0; r < 8; ++r) nacc[r] = 0.f;
        for (int k = 0; k < KNN; ++k) {
            int id = sId[p][k];
#pragma unroll
            for (int r = 0; r < 8; ++r)
                nacc[r] += nfb[(size_t)(oc * 8 + r) * NPTS + id];
        }
#pragma unroll
        for (int r = 0; r < 8; ++r) pk[r] = (__bf16)(nacc[r] * 0.0625f);
        *(v8bf*)(sMT + p * 264 + 128 + oc * 8) = pk;
    }
    __syncthreads();
    // phase 3: WMMA + bias + store + GN stat accumulation
    const int wv = tid >> 5, lane = tid & 31;
    const int mb = (lane >> 4) << 3, j = lane & 15;
    for (int ot = wv; ot < COUT / 16; ot += 8) {
        v8f acc[1];
        wmma_row<256, 1>(PWbf + (size_t)ot * 16 * 256, sMT, lane, acc);
#pragma unroll
        for (int r = 0; r < 8; ++r) {
            int o = ot * 16 + mb + r;
            float v = acc[0][r] + PB[o];
            T[((size_t)b * COUT + o) * NPTS + n0 + j] = v;
            int grp = o >> GSHIFT;
            atomicAdd(&sGS[grp], v);
            atomicAdd(&sGSS[grp], v * v);
        }
    }
    __syncthreads();
    if (tid < 16) { atomicAdd(&gsum[b * 16 + tid], sGS[tid]);
                    atomicAdd(&gssq[b * 16 + tid], sGSS[tid]); }
}

// GN normalize + relu (elementwise second pass), float4-vectorized
__global__ void k_pool_b(const float* __restrict__ T, const float* __restrict__ mu,
                         const float* __restrict__ inv, const float* __restrict__ gw,
                         const float* __restrict__ gb, float* __restrict__ Y,
                         int COUT, int gshift, int total4) {
    int i = blockIdx.x * 256 + threadIdx.x;
    if (i >= total4) return;
    int base = i * 4;
    int o = (base / NPTS) % COUT;       // NPTS % 4 == 0 -> same channel for all 4
    int b = base / (NPTS * COUT);
    int grp = o >> gshift;
    float m = mu[b * 16 + grp], iv = inv[b * 16 + grp];
    float w = gw[o], bb = gb[o];
    float4 t = ((const float4*)T)[i];
    float4 y;
    y.x = (t.x - m) * iv * w + bb; y.x = y.x > 0.f ? y.x : 0.f;
    y.y = (t.y - m) * iv * w + bb; y.y = y.y > 0.f ? y.y : 0.f;
    y.z = (t.z - m) * iv * w + bb; y.z = y.z > 0.f ? y.z : 0.f;
    y.w = (t.w - m) * iv * w + bb; y.w = y.w > 0.f ? y.w : 0.f;
    ((float4*)Y)[i] = y;
}

// ----- shortcut conv (512x128) stats pass: WMMA, accumulate GN sums only ----

__global__ void k_sc_stats(const float* __restrict__ F, const __bf16* __restrict__ SWbf,
                           const float* __restrict__ SB,
                           float* __restrict__ gsum, float* __restrict__ gssq) {
    const int b = blockIdx.y, n0 = blockIdx.x * 64, tid = threadIdx.x;
    __shared__ alignas(16) __bf16 sFT[64 * 136];
    __shared__ float sGS[16], sGSS[16];
    stage_tile<128, 64>(F + (size_t)b * 128 * NPTS, n0, sFT, tid);
    if (tid < 16) { sGS[tid] = 0.f; sGSS[tid] = 0.f; }
    __syncthreads();
    const int wv = tid >> 5, lane = tid & 31, mb = (lane >> 4) << 3;
    for (int ot = wv; ot < 32; ot += 8) {
        v8f acc[4];
        wmma_row<128, 4>(SWbf + (size_t)ot * 16 * 128, sFT, lane, acc);
#pragma unroll
        for (int r = 0; r < 8; ++r) {
            int o = ot * 16 + mb + r;
            int grp = o >> 5;
            float s = 0.f, s2 = 0.f;
#pragma unroll
            for (int p = 0; p < 4; ++p) {
                float v = acc[p][r] + SB[o];
                s += v; s2 += v * v;
            }
            atomicAdd(&sGS[grp], s);
            atomicAdd(&sGSS[grp], s2);
        }
    }
    __syncthreads();
    if (tid < 16) { atomicAdd(&gsum[b * 16 + tid], sGS[tid]);
                    atomicAdd(&gssq[b * 16 + tid], sGSS[tid]); }
}

// ----- final: main = mlp2(512x256)*y2, short = GN(sc(512x128)*F), leaky 0.01

__global__ void k_final(const float* __restrict__ Y2, const __bf16* __restrict__ MWbf,
                        const float* __restrict__ MB, const float* __restrict__ F,
                        const __bf16* __restrict__ SWbf, const float* __restrict__ SB,
                        const float* __restrict__ sgw, const float* __restrict__ sgb,
                        const float* __restrict__ smu, const float* __restrict__ sinv,
                        float* __restrict__ OUT) {
    const int b = blockIdx.y, n0 = blockIdx.x * 64, tid = threadIdx.x;
    __shared__ alignas(16) __bf16 sYT[64 * 264];
    __shared__ alignas(16) __bf16 sFT[64 * 136];
    stage_tile<256, 64>(Y2 + (size_t)b * 256 * NPTS, n0, sYT, tid);
    stage_tile<128, 64>(F  + (size_t)b * 128 * NPTS, n0, sFT, tid);
    __syncthreads();
    const int wv = tid >> 5, lane = tid & 31;
    const int mb = (lane >> 4) << 3, j = lane & 15;
    for (int ot = wv; ot < 32; ot += 8) {
        v8f am[4], as[4];
        wmma_row<256, 4>(MWbf + (size_t)ot * 16 * 256, sYT, lane, am);
        wmma_row<128, 4>(SWbf + (size_t)ot * 16 * 128, sFT, lane, as);
#pragma unroll
        for (int r = 0; r < 8; ++r) {
            int o = ot * 16 + mb + r;
            int grp = o >> 5;
            float m = smu[b * 16 + grp], iv = sinv[b * 16 + grp];
            float w = sgw[o], bb2 = sgb[o];
            float sb = SB[o], mbias = MB[o];
#pragma unroll
            for (int p = 0; p < 4; ++p) {
                float s  = as[p][r] + sb;
                float sn = (s - m) * iv * w + bb2;
                float v  = am[p][r] + mbias + sn;
                OUT[((size_t)b * 512 + o) * NPTS + n0 + p * 16 + j] = v > 0.f ? v : 0.01f * v;
            }
        }
    }
}

// ---------------------------------------------------------------------------

extern "C" void kernel_launch(void* const* d_in, const int* in_sizes, int n_in,
                              void* d_out, int out_size, void* d_ws, size_t ws_size,
                              hipStream_t stream) {
    const float* coords = (const float*)d_in[0];
    const float* feats  = (const float*)d_in[1];
    const float* kdist  = (const float*)d_in[2];
    const int*   kidx   = (const int*)  d_in[3];
    const float* w1  = (const float*)d_in[4];   const float* b1  = (const float*)d_in[5];
    const float* l1w = (const float*)d_in[6];   const float* l1b = (const float*)d_in[7];
    const float* l1gw= (const float*)d_in[8];   const float* l1gb= (const float*)d_in[9];
    const float* p1w = (const float*)d_in[10];  const float* p1b = (const float*)d_in[11];
    const float* p1gw= (const float*)d_in[12];  const float* p1gb= (const float*)d_in[13];
    const float* l2w = (const float*)d_in[14];  const float* l2b = (const float*)d_in[15];
    const float* l2gw= (const float*)d_in[16];  const float* l2gb= (const float*)d_in[17];
    const float* p2w = (const float*)d_in[18];  const float* p2b = (const float*)d_in[19];
    const float* p2gw= (const float*)d_in[20];  const float* p2gb= (const float*)d_in[21];
    const float* mw  = (const float*)d_in[22];  const float* mbb = (const float*)d_in[23];
    const float* scw = (const float*)d_in[24];  const float* scb = (const float*)d_in[25];
    const float* scgw= (const float*)d_in[26];  const float* scgb= (const float*)d_in[27];
    (void)in_sizes; (void)n_in; (void)out_size; (void)ws_size;

    float* ws  = (float*)d_ws;
    float* out = (float*)d_out;

    // stats region (first 1024 floats, zeroed every call)
    float* mom   = ws;          // NB*65
    float* lmu   = ws + 130;    // 2 convs * NB * 16
    float* linv  = ws + 194;
    float* p1sum = ws + 258; float* p1ssq = ws + 290; float* p1mu = ws + 322; float* p1inv = ws + 354;
    float* p2sum = ws + 386; float* p2ssq = ws + 418; float* p2mu = ws + 450; float* p2inv = ws + 482;
    float* scsum = ws + 514; float* scssq = ws + 546; float* scmu = ws + 578; float* scinv = ws + 610;
    // big fp32 buffers
    float* x1 = ws + 1024;                         // NB*128*N
    float* t1 = x1 + (size_t)NB * 128 * NPTS;      // NB*128*N
    float* y1 = t1 + (size_t)NB * 128 * NPTS;      // NB*128*N
    float* t2 = y1 + (size_t)NB * 128 * NPTS;      // NB*256*N
    float* y2 = t2 + (size_t)NB * 256 * NPTS;      // NB*256*N
    float* wend = y2 + (size_t)NB * 256 * NPTS;
    // bf16 weight cache (offsets all 16B aligned)
    __bf16* wb_w1 = (__bf16*)(wend);
    __bf16* wb_p1 = wb_w1 + 128 * 128;
    __bf16* wb_p2 = wb_p1 + 128 * 256;
    __bf16* wb_m  = wb_p2 + 256 * 256;
    __bf16* wb_sc = wb_m  + 512 * 256;

    dim3 g16(NPTS / 16, NB);
    dim3 g64(NPTS / 64, NB);

    k_zero<<<1, 256, 0, stream>>>(ws);
    k_cvt_bf16<<<(128 * 128 / 2 + 255) / 256, 256, 0, stream>>>(w1,  wb_w1, 128 * 128);
    k_cvt_bf16<<<(128 * 256 / 2 + 255) / 256, 256, 0, stream>>>(p1w, wb_p1, 128 * 256);
    k_cvt_bf16<<<(256 * 256 / 2 + 255) / 256, 256, 0, stream>>>(p2w, wb_p2, 256 * 256);
    k_cvt_bf16<<<(512 * 256 / 2 + 255) / 256, 256, 0, stream>>>(mw,  wb_m,  512 * 256);
    k_cvt_bf16<<<(512 * 128 / 2 + 255) / 256, 256, 0, stream>>>(scw, wb_sc, 512 * 128);

    k_conv1<<<g64, 256, 0, stream>>>(feats, wb_w1, b1, x1);
    k_geomom<<<dim3(64, NB), 256, 0, stream>>>(coords, kidx, kdist, mom);
    k_lse_stats<<<1, 64, 0, stream>>>(mom, l1w, l1b, l2w, l2b, lmu, linv);

    k_pool_a<128, 3><<<g16, 256, 0, stream>>>(coords, kidx, kdist,
        l1w, l1b, l1gw, l1gb, lmu, linv, x1, wb_p1, p1b, t1, p1sum, p1ssq);
    k_stat_fin<<<1, 32, 0, stream>>>(p1sum, p1ssq, p1mu, p1inv, 8.f * NPTS);
    int tot1_4 = NB * 128 * NPTS / 4;
    k_pool_b<<<(tot1_4 + 255) / 256, 256, 0, stream>>>(t1, p1mu, p1inv, p1gw, p1gb, y1, 128, 3, tot1_4);

    k_pool_a<256, 4><<<g16, 256, 0, stream>>>(coords, kidx, kdist,
        l2w, l2b, l2gw, l2gb, lmu + 32, linv + 32, y1, wb_p2, p2b, t2, p2sum, p2ssq);
    k_stat_fin<<<1, 32, 0, stream>>>(p2sum, p2ssq, p2mu, p2inv, 16.f * NPTS);
    int tot2_4 = NB * 256 * NPTS / 4;
    k_pool_b<<<(tot2_4 + 255) / 256, 256, 0, stream>>>(t2, p2mu, p2inv, p2gw, p2gb, y2, 256, 4, tot2_4);

    k_sc_stats<<<g64, 256, 0, stream>>>(feats, wb_sc, scb, scsum, scssq);
    k_stat_fin<<<1, 32, 0, stream>>>(scsum, scssq, scmu, scinv, 32.f * NPTS);

    k_final<<<g64, 256, 0, stream>>>(y2, wb_m, mbb, feats, wb_sc, scb, scgw, scgb, scmu, scinv, out);
}
